// DecoderLSTM_26757646254497
// MI455X (gfx1250) — matrix operand order repaired and checked by reading, not compiled, not run on hardware
//
#include <hip/hip_runtime.h>

// ---------- CDNA5 WMMA types ----------
typedef __attribute__((ext_vector_type(16))) __bf16 v16bf;
typedef __attribute__((ext_vector_type(8)))  float  v8f;
typedef __attribute__((ext_vector_type(4)))  int    v4i;

// Problem constants (from reference): B=256, F=64, H=1024, L=2
static constexpr int Bn = 256;
static constexpr int Fn = 64;
static constexpr int Hn = 1024;

// ---------- async global->LDS staging (CDNA5, ASYNCcnt-tracked) ----------
#if __has_builtin(__builtin_amdgcn_global_load_async_to_lds_b128) && \
    __has_builtin(__builtin_amdgcn_s_wait_asynccnt)
#define USE_ASYNC_LDS 1
#else
#define USE_ASYNC_LDS 0
#endif

#if USE_ASYNC_LDS
typedef __attribute__((address_space(1))) v4i gv4i;   // global int4
typedef __attribute__((address_space(3))) v4i lv4i;   // LDS int4
#endif

// Copy 32 bytes (16 bf16) global -> LDS for this thread.
__device__ __forceinline__ void stage32B(const __bf16* g, __bf16* l) {
#if USE_ASYNC_LDS
  v4i* gg = (v4i*)(void*)const_cast<__bf16*>(g);
  v4i* ll = (v4i*)(void*)l;
  gv4i* gp = (gv4i*)gg;   // addrspacecast generic -> global
  lv4i* lp = (lv4i*)ll;   // addrspacecast generic -> LDS
  __builtin_amdgcn_global_load_async_to_lds_b128(gp,     lp,     0, 0);
  __builtin_amdgcn_global_load_async_to_lds_b128(gp + 1, lp + 1, 0, 0);
#else
  const uint4* gs = (const uint4*)g;
  uint4*       ls = (uint4*)l;
  ls[0] = gs[0];
  ls[1] = gs[1];
#endif
}
__device__ __forceinline__ void wait_async0() {
#if USE_ASYNC_LDS
  __builtin_amdgcn_s_wait_asynccnt(0);
#endif
}

// ===================================================================
// Pack fp32 weights (N x K row-major) into a staged-panel layout:
//   idx = (((jt*nchunk + c)*8 + g*2 + p) * 512) + lane*16 + e
//   n = g*gstride + jt*16 + (lane & 15)
//   k = c*32 + (lane>>4)*8 + (e&7) + (e>>3)*16   (WMMA-B 32x16 per-lane layout)
//   p = 0 -> bf16 hi, p = 1 -> bf16 residual lo
// One (jt, c) panel = 8 KB contiguous = exactly one block-wide stage.
// ===================================================================
__global__ void pack_w_kernel(const float* __restrict__ W, __bf16* __restrict__ out,
                              int nchunk, int gstride, int K) {
  long idx = (long)blockIdx.x * blockDim.x + threadIdx.x;
  int e = idx & 15;
  int L = (idx >> 4) & 31;
  int s = (idx >> 9) & 7;        // g*2 + p
  int p = s & 1, g = s >> 1;
  long cc = idx >> 12;           // jt*nchunk + c
  int c  = (int)(cc % nchunk);
  int jt = (int)(cc / nchunk);
  int n = g * gstride + jt * 16 + (L & 15);
  int k = c * 32 + (L >> 4) * 8 + (e & 7) + (e >> 3) * 16;
  float w = W[(long)n * K + k];
  __bf16 h = (__bf16)w;
  out[idx] = p ? (__bf16)(w - (float)h) : h;
}

__global__ void bias_kernel(const float* __restrict__ bi0, const float* __restrict__ bh0,
                            const float* __restrict__ bi1, const float* __restrict__ bh1,
                            float* __restrict__ b0, float* __restrict__ b1) {
  int i = blockIdx.x * blockDim.x + threadIdx.x;
  if (i < 4 * Hn) { b0[i] = bi0[i] + bh0[i]; b1[i] = bi1[i] + bh1[i]; }
}

// ---------- device helpers ----------
__device__ __forceinline__ float sigmoid_(float x) {
  return 1.0f / (1.0f + __expf(-x));
}
__device__ __forceinline__ float tanh_(float x) {
  x = fminf(fmaxf(x, -15.0f), 15.0f);
  float e = __expf(2.0f * x);
  return (e - 1.0f) / (e + 1.0f);
}

// Per-lane fp32 A fragment load + bf16 hi/lo split (A 16x32 layout).
__device__ __forceinline__ void load_a(const float* __restrict__ row, int kbase, int half,
                                       v16bf& ahi, v16bf& alo) {
  const float* p0 = row + kbase + half * 8;
  const float* p1 = row + kbase + 16 + half * 8;
#pragma unroll
  for (int i = 0; i < 8; ++i) {
    float w = p0[i];
    __bf16 h = (__bf16)w;
    ahi[i] = h;
    alo[i] = (__bf16)(w - (float)h);
  }
#pragma unroll
  for (int i = 0; i < 8; ++i) {
    float w = p1[i];
    __bf16 h = (__bf16)w;
    ahi[8 + i] = h;
    alo[8 + i] = (__bf16)(w - (float)h);
  }
}

// Split-precision product: acc += Ahi*Bhi + Ahi*Blo + Alo*Bhi  (fp32 accumulate)
__device__ __forceinline__ v8f wmma3(v16bf ahi, v16bf alo, v16bf bhi, v16bf blo, v8f acc) {
  acc = __builtin_amdgcn_wmma_f32_16x16x32_bf16(false, ahi, false, bhi, (short)0, acc, false, false);
  acc = __builtin_amdgcn_wmma_f32_16x16x32_bf16(false, ahi, false, blo, (short)0, acc, false, false);
  acc = __builtin_amdgcn_wmma_f32_16x16x32_bf16(false, alo, false, bhi, (short)0, acc, false, false);
  return acc;
}

// ===================================================================
// One GEMM phase: acc[g] += A(16 x 32*nchunk) @ Wpanel_g^T for 4 tiles g.
// All 8 waves of the block share the same B panel, staged chunk-by-chunk
// into LDS (double buffered) with async global->LDS loads.
// sB: __shared__ 2 x 4096 bf16 (2 x 8 KB).
// ===================================================================
__device__ __forceinline__ void gemm_phase(const float* __restrict__ arow, int nchunk,
                                           const __bf16* __restrict__ panel,  // + jt*nchunk*4096
                                           __bf16* sB, int tid, int lane, int half,
                                           v8f acc[4]) {
  stage32B(panel + tid * 16, sB + tid * 16);            // chunk 0 -> buf 0
  for (int c = 0; c < nchunk; ++c) {
    wait_async0();
    __syncthreads();                                    // chunk c resident for all waves
    if (c + 1 < nchunk)
      stage32B(panel + (long)(c + 1) * 4096 + tid * 16,
               sB + ((c + 1) & 1) * 4096 + tid * 16);   // prefetch next chunk
    v16bf ahi, alo;
    load_a(arow, c * 32, half, ahi, alo);
    const __bf16* sb = sB + (c & 1) * 4096;
#pragma unroll
    for (int g = 0; g < 4; ++g) {
      v16bf bhi = *(const v16bf*)(sb + (g * 2 + 0) * 512 + lane * 16);
      v16bf blo = *(const v16bf*)(sb + (g * 2 + 1) * 512 + lane * 16);
      acc[g] = wmma3(ahi, alo, bhi, blo, acc[g]);
    }
  }
  __syncthreads();                                      // protect buffers across phases
}

// ===================================================================
// One LSTM layer step: gates = A0@W0^T + A1@W1^T + bias; cell update.
// Wave: 1 batch tile x 16 hidden cols x all 4 gates (i,f,g,o in regs).
// Block: 8 waves sharing jt (same B panel). Grid: 128 x 256.
// ===================================================================
__global__ void __launch_bounds__(256)
lstm_layer_kernel(const float* __restrict__ A0, long strideA0, int nchunk0,
                  const __bf16* __restrict__ W0pk,
                  const float* __restrict__ A1, int nchunk1,
                  const __bf16* __restrict__ W1pk,
                  const float* __restrict__ bias,
                  float* __restrict__ c_state, float* __restrict__ h_out) {
  __shared__ __bf16 sB[2 * 4096];
  const int tid  = threadIdx.x;
  const int lane = tid & 31;
  const int wid  = (blockIdx.x * blockDim.x + tid) >> 5;  // 0..1023
  const int mt   = wid & 15;   // batch tile
  const int jt   = wid >> 4;   // hidden-column tile (uniform within block)
  const int half = lane >> 4;
  const int lc   = lane & 15;

  v8f acc[4];
#pragma unroll
  for (int g = 0; g < 4; ++g) {
    float bv = bias[g * Hn + jt * 16 + lc];
#pragma unroll
    for (int r = 0; r < 8; ++r) acc[g][r] = bv;
  }

  const float* arow0 = A0 + (long)(mt * 16 + lc) * strideA0;
  gemm_phase(arow0, nchunk0, W0pk + (long)jt * nchunk0 * 4096, sB, tid, lane, half, acc);
  const float* arow1 = A1 + (long)(mt * 16 + lc) * Hn;
  gemm_phase(arow1, nchunk1, W1pk + (long)jt * nchunk1 * 4096, sB, tid, lane, half, acc);

  // LSTM cell: C/D layout -> VGPR r, lane: M = r + half*8, N = lc
  const int j = jt * 16 + lc;
#pragma unroll
  for (int r = 0; r < 8; ++r) {
    int b = mt * 16 + r + half * 8;
    long idx = (long)b * Hn + j;
    float iv = sigmoid_(acc[0][r]);
    float fv = sigmoid_(acc[1][r]);
    float gv = tanh_(acc[2][r]);
    float ov = sigmoid_(acc[3][r]);
    float cn = fv * c_state[idx] + iv * gv;
    c_state[idx] = cn;
    h_out[idx]   = ov * tanh_(cn);
  }
}

// ===================================================================
// pred = h1 @ W_fc^T + b_fc  (256 x 64, K=1024) -> d_out[:, t, :]
// 2 blocks x 8 waves; wave = batch tile; 4 N-tiles as the "gates".
// ===================================================================
__global__ void __launch_bounds__(256)
fc_kernel(const float* __restrict__ A, const __bf16* __restrict__ Wpk,
          const float* __restrict__ bfc, float* __restrict__ out, int t, int OW) {
  __shared__ __bf16 sB[2 * 4096];
  const int tid  = threadIdx.x;
  const int lane = tid & 31;
  const int mt   = (blockIdx.x * blockDim.x + tid) >> 5;  // 0..15
  const int half = lane >> 4;
  const int lc   = lane & 15;

  v8f acc[4];
#pragma unroll
  for (int g = 0; g < 4; ++g) {
    float bv = bfc[g * 16 + lc];
#pragma unroll
    for (int r = 0; r < 8; ++r) acc[g][r] = bv;
  }

  const float* arow = A + (long)(mt * 16 + lc) * Hn;
  gemm_phase(arow, 32, Wpk, sB, tid, lane, half, acc);

#pragma unroll
  for (int g = 0; g < 4; ++g) {
#pragma unroll
    for (int r = 0; r < 8; ++r) {
      int b = mt * 16 + r + half * 8;
      out[(long)b * OW * Fn + (long)t * Fn + g * 16 + lc] = acc[g][r];
    }
  }
}

// ===================================================================
// Host driver
// ===================================================================
extern "C" void kernel_launch(void* const* d_in, const int* in_sizes, int n_in,
                              void* d_out, int out_size, void* d_ws, size_t ws_size,
                              hipStream_t stream) {
  (void)in_sizes; (void)n_in; (void)ws_size;
  const float* x0   = (const float*)d_in[0];   // (256,1,64)
  const float* hid  = (const float*)d_in[1];   // (2,256,1024)
  const float* cel  = (const float*)d_in[2];   // (2,256,1024)
  const float* Wih0 = (const float*)d_in[3];   // (4096,64)
  const float* Whh0 = (const float*)d_in[4];   // (4096,1024)
  const float* bih0 = (const float*)d_in[5];
  const float* bhh0 = (const float*)d_in[6];
  const float* Wih1 = (const float*)d_in[7];   // (4096,1024)
  const float* Whh1 = (const float*)d_in[8];   // (4096,1024)
  const float* bih1 = (const float*)d_in[9];
  const float* bhh1 = (const float*)d_in[10];
  const float* Wfc  = (const float*)d_in[11];  // (64,1024)
  const float* bfc  = (const float*)d_in[12];
  float* out = (float*)d_out;                  // (256, OW, 64)
  const int OW = out_size / (Bn * Fn);         // 96

  // ---- workspace layout ----
  char* ws = (char*)d_ws;
  size_t off = 0;
  auto alloc = [&](size_t bytes) -> char* {
    char* p = ws + off;
    off += (bytes + 255) & ~(size_t)255;
    return p;
  };
  const size_t E_ih0 = (size_t)4096 * 64 * 2;    // packed elems (hi+lo)
  const size_t E_hh  = (size_t)4096 * 1024 * 2;
  const size_t E_fc  = (size_t)64 * 1024 * 2;
  __bf16* pih0 = (__bf16*)alloc(E_ih0 * 2);
  __bf16* phh0 = (__bf16*)alloc(E_hh * 2);
  __bf16* pih1 = (__bf16*)alloc(E_hh * 2);
  __bf16* phh1 = (__bf16*)alloc(E_hh * 2);
  __bf16* pfc  = (__bf16*)alloc(E_fc * 2);
  float* b0 = (float*)alloc(4096 * 4);
  float* b1 = (float*)alloc(4096 * 4);
  const size_t STATE = (size_t)Bn * Hn * 4;      // 1 MB
  float* h0buf[2] = { (float*)alloc(STATE), (float*)alloc(STATE) };
  float* h1buf[2] = { (float*)alloc(STATE), (float*)alloc(STATE) };
  float* c0 = (float*)alloc(STATE);
  float* c1 = (float*)alloc(STATE);

  // ---- repack weights into staged-panel bf16 hi/lo layout ----
  pack_w_kernel<<<(int)(E_ih0 / 256), 256, 0, stream>>>(Wih0, pih0, 2,  1024, 64);
  pack_w_kernel<<<(int)(E_hh  / 256), 256, 0, stream>>>(Whh0, phh0, 32, 1024, 1024);
  pack_w_kernel<<<(int)(E_hh  / 256), 256, 0, stream>>>(Wih1, pih1, 32, 1024, 1024);
  pack_w_kernel<<<(int)(E_hh  / 256), 256, 0, stream>>>(Whh1, phh1, 32, 1024, 1024);
  pack_w_kernel<<<(int)(E_fc  / 256), 256, 0, stream>>>(Wfc,  pfc,  32, 16,   1024);
  bias_kernel<<<16, 256, 0, stream>>>(bih0, bhh0, bih1, bhh1, b0, b1);

  // ---- initialize recurrent state ----
  (void)hipMemcpyAsync(h0buf[0], hid,           STATE, hipMemcpyDeviceToDevice, stream);
  (void)hipMemcpyAsync(h1buf[0], hid + Bn * Hn, STATE, hipMemcpyDeviceToDevice, stream);
  (void)hipMemcpyAsync(c0,       cel,           STATE, hipMemcpyDeviceToDevice, stream);
  (void)hipMemcpyAsync(c1,       cel + Bn * Hn, STATE, hipMemcpyDeviceToDevice, stream);

  // ---- time loop: 3 kernels per step ----
  for (int t = 0; t < OW; ++t) {
    const int cur = t & 1, nxt = cur ^ 1;
    const float* a0 = (t == 0) ? x0 : (out + (size_t)(t - 1) * Fn);
    const long  s0  = (t == 0) ? (long)Fn : (long)OW * Fn;
    lstm_layer_kernel<<<128, 256, 0, stream>>>(
        a0, s0, Fn / 32, pih0,
        h0buf[cur], Hn / 32, phh0,
        b0, c0, h0buf[nxt]);
    lstm_layer_kernel<<<128, 256, 0, stream>>>(
        h0buf[nxt], (long)Hn, Hn / 32, pih1,
        h1buf[cur], Hn / 32, phh1,
        b1, c1, h1buf[nxt]);
    fc_kernel<<<2, 256, 0, stream>>>(h1buf[nxt], pfc, bfc, out, t, OW);
  }
}
